// ConditionalRandomField_6536940224631
// MI455X (gfx1250) — matrix operand order repaired and checked by reading, not compiled, not run on hardware
//
#include <hip/hip_runtime.h>
#include <math.h>

// Problem constants (from the reference)
#define Bc 256
#define Tc 512
#define Nc 256
#define STARTc 254
#define STOPc  255

#define ROWS_PER_BLOCK 16          // batch rows per workgroup
#define WAVES 16                   // 16 waves * 32 lanes = 512 threads
#define EMPAD 260                  // f32 emit row stride (pad: conflict-free b32 access)
#define EAPADH 264                 // f16 expA row stride in halves: 528B/row -> exactly
                                   // 2-way bank replay on ds_load_b128 (the b128 minimum)

typedef _Float16 v8h  __attribute__((ext_vector_type(8)));
typedef _Float16 v16h __attribute__((ext_vector_type(16)));
typedef float    v8f  __attribute__((ext_vector_type(8)));

// ---------------------------------------------------------------------------
// Kernel 1: CRF forward scan (log-denominator) — one block per 16 batch rows.
// Per step t: S = expA(f16) @ [expT_hi + expT_lo](f16)^T via v_wmma_f32_16x16x32_f16
// (two passes kill the systematic f16 bias of the constant transition matrix);
// alpha_new = emit + M + log(S); renormalize expA = exp(alpha_new - M_new).
// expT slices register-resident; expA double-buffered in LDS as f16.
// ---------------------------------------------------------------------------
__global__ __launch_bounds__(512)
void crf_forward_scan(const float* __restrict__ inputs,      // [B, T, N]
                      const float* __restrict__ trans,       // [N, N]
                      const int*   __restrict__ mask,        // [B, T]
                      float*       __restrict__ denom)       // [B]
{
  __shared__ _Float16 sExpA[2][ROWS_PER_BLOCK][EAPADH];  // exp(alpha - Mrow), f16, dbl-buffered
  __shared__ float    sEmit[ROWS_PER_BLOCK][EMPAD];      // emissions for current t
  __shared__ float    sWaveMax[WAVES][ROWS_PER_BLOCK];   // per-wave row maxima
  __shared__ float    sMrow[ROWS_PER_BLOCK];             // running row max M_b
  __shared__ float    sRowSum[2][ROWS_PER_BLOCK];        // sum_j expA (mask==0 path)
  __shared__ int      sMask[ROWS_PER_BLOCK];

  const int tid  = threadIdx.x;
  const int wave = tid >> 5;       // 0..15 : owns output tag columns [16w, 16w+16)
  const int lane = tid & 31;
  const int l15  = lane & 15;
  const int hi   = lane >> 4;      // 0/1 half-wave (K-split in 16-bit WMMA layouts)
  const int b0   = blockIdx.x * ROWS_PER_BLOCK;
  const int nbase = wave * 16;

  // ---- Prologue: register-resident B = exp(trans)^T slice (256 x 16), f16 hi+lo.
  // 16-bit B 32x16 layout (matches A): elements 0-7: K = 8*hi + 0..7,
  // elements 8-15: K = 16 + 8*hi + 0..7; column = nbase + l15.
  v16h Bh[8], Bl[8];
  {
    const float* trow = trans + (size_t)(nbase + l15) * Nc;   // row i = tag column
    #pragma unroll
    for (int s = 0; s < 8; ++s) {
      #pragma unroll
      for (int e = 0; e < 16; ++e) {
        float v = __expf(trow[32 * s + (e < 8 ? 8 * hi + e : 16 + 8 * hi + (e - 8))]);
        _Float16 h = (_Float16)v;
        Bh[s][e] = h;
        Bl[s][e] = (_Float16)(v - (float)h);   // residual: cancels constant-matrix bias
      }
    }
  }

  // ---- Init: alpha0[b,i] = trans[i, START] + inputs[b,0,i]; wave w -> row w
  {
    const int m = wave;
    const float* inr = inputs + (size_t)(b0 + m) * Tc * Nc;  // t = 0
    float vals[8];
    float lmax = -INFINITY;
    #pragma unroll
    for (int q = 0; q < 8; ++q) {
      int i = lane + 32 * q;
      float v = trans[(size_t)i * Nc + STARTc] + inr[i];
      vals[q] = v;
      lmax = fmaxf(lmax, v);
    }
    #pragma unroll
    for (int off = 16; off; off >>= 1) lmax = fmaxf(lmax, __shfl_xor(lmax, off, 32));
    float lsum = 0.f;
    #pragma unroll
    for (int q = 0; q < 8; ++q) {
      int i = lane + 32 * q;
      float e = __expf(vals[q] - lmax);
      sExpA[0][m][i] = (_Float16)e;
      lsum += e;
    }
    #pragma unroll
    for (int off = 16; off; off >>= 1) lsum += __shfl_xor(lsum, off, 32);
    if (lane == 0) { sMrow[m] = lmax; sRowSum[0][m] = lsum; }
  }
  __syncthreads();

  // ---- Scan over t = 1 .. T-1 (fully on-chip; no global sync)
  for (int t = 1; t < Tc; ++t) {
    const int cur = (t - 1) & 1;
    const int nxt = t & 1;

    // Stage emissions for step t (coalesced) + mask column
    #pragma unroll
    for (int q = 0; q < 8; ++q) {
      int e   = tid + 512 * q;
      int row = e >> 8;
      int col = e & 255;
      sEmit[row][col] = inputs[(size_t)(b0 + row) * Tc * Nc + (size_t)t * Nc + col];
    }
    if (tid < ROWS_PER_BLOCK) sMask[tid] = mask[(size_t)(b0 + tid) * Tc + t];
    __syncthreads();   // emits ready AND previous step's expA[cur] writes visible

    // GEMM: S(16x16) = expA[cur](16x256,f16) @ B(256x16), K in slices of 32.
    // 16-bit A 16x32 layout: lane row = l15; VGPRs 0-3: K = 8*hi + 0..7,
    // VGPRs 4-7: K = 16 + 8*hi + 0..7  -> two contiguous 8-half (16B) chunks.
    v8f acc = {0.f, 0.f, 0.f, 0.f, 0.f, 0.f, 0.f, 0.f};
    {
      const _Float16* aptr = &sExpA[cur][l15][0];
      #pragma unroll
      for (int s = 0; s < 8; ++s) {
        v8h c0 = *(const v8h*)(aptr + 32 * s + 8 * hi);        // ds_load_b128
        v8h c1 = *(const v8h*)(aptr + 32 * s + 16 + 8 * hi);   // ds_load_b128
        v16h a = __builtin_shufflevector(c0, c1,
                  0, 1, 2, 3, 4, 5, 6, 7, 8, 9, 10, 11, 12, 13, 14, 15);
        acc = __builtin_amdgcn_wmma_f32_16x16x32_f16(
            false, a, false, Bh[s], (short)0, acc, false, false);
        acc = __builtin_amdgcn_wmma_f32_16x16x32_f16(
            false, a, false, Bl[s], (short)0, acc, false, false);
      }
    }

    // Finalize: alpha_new = emit + Mprev + log(S)   (mask==0: logsumexp_j alpha)
    float aval[8];
    #pragma unroll
    for (int r = 0; r < 8; ++r) {
      int m = r + 8 * hi;
      float Mp = sMrow[m];
      aval[r] = sMask[m] ? (sEmit[m][nbase + l15] + Mp + __logf(acc[r]))
                         : (Mp + __logf(sRowSum[cur][m]));
    }

    // Per-wave row maxima over this wave's 16 tag columns
    #pragma unroll
    for (int r = 0; r < 8; ++r) {
      float mx = aval[r];
      mx = fmaxf(mx, __shfl_xor(mx, 1, 32));
      mx = fmaxf(mx, __shfl_xor(mx, 2, 32));
      mx = fmaxf(mx, __shfl_xor(mx, 4, 32));
      mx = fmaxf(mx, __shfl_xor(mx, 8, 32));
      if (l15 == 0) sWaveMax[wave][r + 8 * hi] = mx;
    }
    __syncthreads();

    // New running max per row; zero next rowsum
    if (tid < ROWS_PER_BLOCK) {
      float mx = sWaveMax[0][tid];
      #pragma unroll
      for (int w = 1; w < WAVES; ++w) mx = fmaxf(mx, sWaveMax[w][tid]);
      sMrow[tid] = mx;
      sRowSum[nxt][tid] = 0.f;
    }
    __syncthreads();

    // Renormalize into expA[nxt] (f16) + accumulate row sums (f32)
    #pragma unroll
    for (int r = 0; r < 8; ++r) {
      int m = r + 8 * hi;
      float e = __expf(aval[r] - sMrow[m]);
      sExpA[nxt][m][nbase + l15] = (_Float16)e;
      float sm = e;
      sm += __shfl_xor(sm, 1, 32);
      sm += __shfl_xor(sm, 2, 32);
      sm += __shfl_xor(sm, 4, 32);
      sm += __shfl_xor(sm, 8, 32);
      if (l15 == 0) atomicAdd(&sRowSum[nxt][m], sm);
    }
    __syncthreads();
  }

  // ---- Epilogue: denom[b] = M_b + log( sum_j expA[b,j] * exp(trans[STOP, j]) )
  {
    const int m = wave;               // wave w -> row w; last buffer is (Tc-1)&1 == 1
    float lsum = 0.f;
    #pragma unroll
    for (int q = 0; q < 8; ++q) {
      int j = lane + 32 * q;
      lsum += (float)sExpA[1][m][j] * __expf(trans[(size_t)STOPc * Nc + j]);
    }
    #pragma unroll
    for (int off = 16; off; off >>= 1) lsum += __shfl_xor(lsum, off, 32);
    if (lane == 0) denom[b0 + m] = sMrow[m] + __logf(lsum);
  }
}

// ---------------------------------------------------------------------------
// Kernel 2: joint (numerator) score per batch row + final scalar reduction.
// ---------------------------------------------------------------------------
__global__ __launch_bounds__(256)
void crf_joint_reduce(const float* __restrict__ inputs,
                      const float* __restrict__ trans,
                      const int*   __restrict__ tags,
                      const int*   __restrict__ mask,
                      const float* __restrict__ denom,
                      float*       __restrict__ out)
{
  int b = blockIdx.x * blockDim.x + threadIdx.x;
  if (b >= Bc) return;
  const int*   tg  = tags  + (size_t)b * Tc;
  const int*   mk  = mask  + (size_t)b * Tc;
  const float* inp = inputs + (size_t)b * Tc * Nc;

  float score   = trans[(size_t)tg[0] * Nc + STARTc];
  float emitsum = 0.f;
  int   prev    = tg[0];
  int   msum    = mk[0];
  for (int t = 1; t < Tc; ++t) {
    int cu = tg[t];
    score   += trans[(size_t)cu * Nc + prev] * (float)mk[t];
    emitsum += inp[(size_t)(t - 1) * Nc + tg[t - 1]] * (float)mk[t - 1];
    prev = cu;
    msum += mk[t];
  }
  score += emitsum;

  int last_idx = msum - 1;
  int last_tag = tg[last_idx];
  score += trans[(size_t)STOPc * Nc + last_tag]
         + inp[(size_t)(Tc - 1) * Nc + last_tag] * (float)mk[Tc - 1];

  atomicAdd(out, score - denom[b]);
}

// ---------------------------------------------------------------------------
extern "C" void kernel_launch(void* const* d_in, const int* in_sizes, int n_in,
                              void* d_out, int out_size, void* d_ws, size_t ws_size,
                              hipStream_t stream) {
  (void)in_sizes; (void)n_in; (void)out_size; (void)ws_size;
  const float* inputs = (const float*)d_in[0];   // [B,T,N] f32
  const float* trans  = (const float*)d_in[1];   // [N,N]   f32
  const int*   tags   = (const int*)d_in[2];     // [B,T]   i32
  const int*   mask   = (const int*)d_in[3];     // [B,T]   i32
  float* out   = (float*)d_out;                  // scalar f32
  float* denom = (float*)d_ws;                   // [B] f32 scratch

  hipMemsetAsync(out, 0, sizeof(float), stream);
  crf_forward_scan<<<Bc / ROWS_PER_BLOCK, WAVES * 32, 0, stream>>>(inputs, trans, mask, denom);
  crf_joint_reduce<<<1, 256, 0, stream>>>(inputs, trans, tags, mask, denom, out);
}
